// Grok1MoE_34789235098063
// MI455X (gfx1250) — matrix operand order repaired and checked by reading, not compiled, not run on hardware
//
#include <hip/hip_runtime.h>
#include <hip/hip_bf16.h>
#include <math.h>

// ---------------------------------------------------------------------------
// Grok1 MoE, sparse top-2 dispatch, bf16 WMMA (v_wmma_f32_16x16x32_bf16),
// LDS-tiled GEMMs (BM=128, BN=64, BK=32), async global->LDS for bf16 operand.
// T=2048 H=2048 E=8 I=4096 k=2
// ---------------------------------------------------------------------------

#define T_NUM 2048
#define H_DIM 2048
#define E_NUM 8
#define I_DIM 4096
#define TOPK  2
#define SOFTCAP 30.0f

#define BM 128
#define BN 64
#define BK 32
#define KPAD 40   // bf16 elements per LDS row (stride); 40*2B=80B keeps 16B align

typedef __attribute__((ext_vector_type(16))) __bf16 v16bf;
typedef __attribute__((ext_vector_type(8)))  float  v8f;

// exact pointee type the async-LDS builtin expects: gcc-vector of 4 ints
typedef int v4i_vs __attribute__((vector_size(16)));
typedef __attribute__((address_space(1))) v4i_vs* as1_v4i;
typedef __attribute__((address_space(3))) v4i_vs* as3_v4i;

#if defined(__HIP_DEVICE_COMPILE__) && __has_builtin(__builtin_amdgcn_global_load_async_to_lds_b128)
#define USE_ASYNC_LDS 1
#endif

// ------------------------------ router -------------------------------------
__global__ __launch_bounds__(256)
void moe_router(const float* __restrict__ hs, const float* __restrict__ gate_w,
                int* __restrict__ counts, int* __restrict__ token_list,
                int* __restrict__ tk_expert, int* __restrict__ tk_pos,
                float* __restrict__ tk_w)
{
    const int wave = threadIdx.x >> 5;
    const int lane = threadIdx.x & 31;
    const int t = blockIdx.x * 8 + wave;

    float acc[E_NUM];
#pragma unroll
    for (int e = 0; e < E_NUM; ++e) acc[e] = 0.0f;

    const float* row = hs + (size_t)t * H_DIM;
    for (int k = lane; k < H_DIM; k += 32) {
        const float x = row[k];
        const float* g = gate_w + (size_t)k * E_NUM;
#pragma unroll
        for (int e = 0; e < E_NUM; ++e) acc[e] += x * g[e];
    }
#pragma unroll
    for (int off = 16; off > 0; off >>= 1) {
#pragma unroll
        for (int e = 0; e < E_NUM; ++e) acc[e] += __shfl_xor(acc[e], off, 32);
    }

    if (lane == 0) {
        float logits[E_NUM];
        float mx = -1e30f;
#pragma unroll
        for (int e = 0; e < E_NUM; ++e) {
            logits[e] = SOFTCAP * tanhf(acc[e] / SOFTCAP);
            mx = fmaxf(mx, logits[e]);
        }
        float denom = 0.0f;
#pragma unroll
        for (int e = 0; e < E_NUM; ++e) denom += __expf(logits[e] - mx);

        int i0 = 0; float b0 = logits[0];
#pragma unroll
        for (int e = 1; e < E_NUM; ++e)
            if (logits[e] > b0) { b0 = logits[e]; i0 = e; }
        int i1 = -1; float b1 = -1e30f;
#pragma unroll
        for (int e = 0; e < E_NUM; ++e)
            if (e != i0 && logits[e] > b1) { b1 = logits[e]; i1 = e; }

        const float w0 = __expf(logits[i0] - mx) / denom;
        const float w1 = __expf(logits[i1] - mx) / denom;

        const int p0 = atomicAdd(&counts[i0], 1);
        const int p1 = atomicAdd(&counts[i1], 1);
        token_list[i0 * T_NUM + p0] = t;
        token_list[i1 * T_NUM + p1] = t;
        tk_expert[t * TOPK + 0] = i0;  tk_expert[t * TOPK + 1] = i1;
        tk_pos[t * TOPK + 0]    = p0;  tk_pos[t * TOPK + 1]    = p1;
        tk_w[t * TOPK + 0]      = w0;  tk_w[t * TOPK + 1]      = w1;
    }
}

// --------------------------- exclusive scan over E --------------------------
__global__ void moe_offsets(const int* __restrict__ counts, int* __restrict__ offsets)
{
    if (threadIdx.x == 0 && blockIdx.x == 0) {
        int s = 0;
        for (int e = 0; e < E_NUM; ++e) { offsets[e] = s; s += counts[e]; }
        offsets[E_NUM] = s;
    }
}

// ------------------------------ helpers -------------------------------------
// Fragment layouts (wave32, v_wmma_f32_16x16x32_bf16):
//  A (16x32): lane L -> row M=L&15, half h=L>>4; elems j: K=16h+j
//             -> one contiguous 32B chunk per lane, loaded as two 16B halves.
//  B (32x16): lane L -> col N=L&15, half h=L>>4; elems j: K=16h+j
//             -> one contiguous 32B chunk per lane when stored K-contiguous.
//  D (16x16 f32): vgpr r, lane L -> row M=r+8*(L>>4), col N=L&15.

__device__ __forceinline__ v16bf frag_ld(const __bf16* p0, const __bf16* p1)
{
    union { uint4 q[2]; v16bf v; } u;
    u.q[0] = *(const uint4*)p0;
    u.q[1] = *(const uint4*)p1;
    return u.v;
}

__device__ __forceinline__ void cvt_store8(__bf16* dst, float4 a, float4 b)
{
    union { __bf16 h[8]; uint4 q; } u;
    u.h[0] = (__bf16)a.x; u.h[1] = (__bf16)a.y;
    u.h[2] = (__bf16)a.z; u.h[3] = (__bf16)a.w;
    u.h[4] = (__bf16)b.x; u.h[5] = (__bf16)b.y;
    u.h[6] = (__bf16)b.z; u.h[7] = (__bf16)b.w;
    *(uint4*)dst = u.q;
}

__device__ __forceinline__ float gelu_tanh(float g)
{
    return 0.5f * g * (1.0f + tanhf(0.7978845608028654f * (g + 0.044715f * g * g * g)));
}

// --------------------------- GEMM1: gate/up + GELU ---------------------------
// Block 256 = 8 waves in 4(M) x 2(N); wave macro-tile 32x32 (2x2 WMMA tiles).
// Grid: (I/BN, ceil(T/BM), E).
__global__ __launch_bounds__(256)
void moe_gemm1(const float* __restrict__ hs, const float* __restrict__ w_gate,
               const float* __restrict__ w_up, const int* __restrict__ counts,
               const int* __restrict__ offsets, const int* __restrict__ token_list,
               __bf16* __restrict__ act)
{
    const int e = blockIdx.z;
    const int cnt = counts[e];
    const int m0 = blockIdx.y * BM;
    if (m0 >= cnt) return;
    const int n0 = blockIdx.x * BN;

    __shared__ __align__(16) __bf16 As[BM * KPAD];
    __shared__ __align__(16) __bf16 Bg[BN * KPAD];
    __shared__ __align__(16) __bf16 Bu[BN * KPAD];

    const int tid = threadIdx.x;
    const int wave = tid >> 5;
    const int lane = tid & 31;
    const int l = lane & 15, hh = lane >> 4;
    const int waveM = wave >> 1;   // 0..3 -> 32-row slab
    const int waveN = wave & 1;    // 0..1 -> 32-col slab

    // loader assignments (hoisted)
    const int ar = tid >> 1;                  // A row 0..127
    const int ak = (tid & 1) * 16;            // A k-offset 0 or 16
    const int grow = (m0 + ar) < cnt ? (m0 + ar) : (cnt - 1);
    const int tok = token_list[e * T_NUM + grow];
    const float* __restrict__ asrc = hs + (size_t)tok * H_DIM + ak;

    const int bk = tid >> 3;                  // B k-row 0..31
    const int bc = (tid & 7) * 8;             // B col group (8 cols)
    const float* __restrict__ gsrc =
        w_gate + (size_t)e * H_DIM * I_DIM + (size_t)bk * I_DIM + n0 + bc;
    const float* __restrict__ usrc =
        w_up   + (size_t)e * H_DIM * I_DIM + (size_t)bk * I_DIM + n0 + bc;

    v8f accg[2][2] = {{{}, {}}, {{}, {}}};
    v8f accu[2][2] = {{{}, {}}, {{}, {}}};

    for (int k0 = 0; k0 < H_DIM; k0 += BK) {
        // ---- stage A (128x32 f32 -> bf16) ----
        {
            float4 f0 = *(const float4*)(asrc + k0 + 0);
            float4 f1 = *(const float4*)(asrc + k0 + 4);
            float4 f2 = *(const float4*)(asrc + k0 + 8);
            float4 f3 = *(const float4*)(asrc + k0 + 12);
            __bf16* d = As + ar * KPAD + ak;
            cvt_store8(d, f0, f1);
            cvt_store8(d + 8, f2, f3);
        }
        // ---- stage B gate & up (32x64 f32 -> bf16, transposed [col][k]) ----
        {
            const size_t koff = (size_t)k0 * I_DIM;
            float4 g0 = *(const float4*)(gsrc + koff);
            float4 g1 = *(const float4*)(gsrc + koff + 4);
            float4 u0 = *(const float4*)(usrc + koff);
            float4 u1 = *(const float4*)(usrc + koff + 4);
            // prefetch next k-panel of weights
            if (k0 + BK < H_DIM) {
                __builtin_prefetch(gsrc + koff + (size_t)BK * I_DIM, 0, 1);
                __builtin_prefetch(usrc + koff + (size_t)BK * I_DIM, 0, 1);
            }
            __bf16* dg = Bg + bc * KPAD + bk;
            __bf16* du = Bu + bc * KPAD + bk;
            dg[0 * KPAD] = (__bf16)g0.x; dg[1 * KPAD] = (__bf16)g0.y;
            dg[2 * KPAD] = (__bf16)g0.z; dg[3 * KPAD] = (__bf16)g0.w;
            dg[4 * KPAD] = (__bf16)g1.x; dg[5 * KPAD] = (__bf16)g1.y;
            dg[6 * KPAD] = (__bf16)g1.z; dg[7 * KPAD] = (__bf16)g1.w;
            du[0 * KPAD] = (__bf16)u0.x; du[1 * KPAD] = (__bf16)u0.y;
            du[2 * KPAD] = (__bf16)u0.z; du[3 * KPAD] = (__bf16)u0.w;
            du[4 * KPAD] = (__bf16)u1.x; du[5 * KPAD] = (__bf16)u1.y;
            du[6 * KPAD] = (__bf16)u1.z; du[7 * KPAD] = (__bf16)u1.w;
        }
        __syncthreads();

        // ---- compute: 2x2 WMMA tiles per wave, gate + up ----
        v16bf af[2];
#pragma unroll
        for (int m = 0; m < 2; ++m) {
            const __bf16* p = As + (waveM * 32 + m * 16 + l) * KPAD + 16 * hh;
            af[m] = frag_ld(p, p + 8);
        }
#pragma unroll
        for (int n = 0; n < 2; ++n) {
            const __bf16* pg = Bg + (waveN * 32 + n * 16 + l) * KPAD + 16 * hh;
            const __bf16* pu = Bu + (waveN * 32 + n * 16 + l) * KPAD + 16 * hh;
            const v16bf bgf = frag_ld(pg, pg + 8);
            const v16bf buf = frag_ld(pu, pu + 8);
#pragma unroll
            for (int m = 0; m < 2; ++m) {
                accg[m][n] = __builtin_amdgcn_wmma_f32_16x16x32_bf16(
                    false, af[m], false, bgf, (short)0, accg[m][n], false, false);
                accu[m][n] = __builtin_amdgcn_wmma_f32_16x16x32_bf16(
                    false, af[m], false, buf, (short)0, accu[m][n], false, false);
            }
        }
        __syncthreads();
    }

    // ---- epilogue: act = gelu(g) * u (bf16) ----
    const int base = offsets[e] + m0;
#pragma unroll
    for (int m = 0; m < 2; ++m) {
#pragma unroll
        for (int n = 0; n < 2; ++n) {
            const int col = n0 + waveN * 32 + n * 16 + l;
#pragma unroll
            for (int r = 0; r < 8; ++r) {
                const int row = waveM * 32 + m * 16 + r + 8 * hh;
                if (m0 + row < cnt) {
                    const float g = accg[m][n][r];
                    const float u = accu[m][n][r];
                    act[(size_t)(base + row) * I_DIM + col] = (__bf16)(gelu_tanh(g) * u);
                }
            }
        }
    }
}

// ------------------------------ GEMM2: down ---------------------------------
// oute[slot, h] = act[slot, :] @ w_down[e];  A operand is bf16 -> async->LDS.
__global__ __launch_bounds__(256)
void moe_gemm2(const __bf16* __restrict__ act, const float* __restrict__ w_down,
               const int* __restrict__ counts, const int* __restrict__ offsets,
               float* __restrict__ oute)
{
    const int e = blockIdx.z;
    const int cnt = counts[e];
    const int m0 = blockIdx.y * BM;
    if (m0 >= cnt) return;
    const int n0 = blockIdx.x * BN;

    __shared__ __align__(16) __bf16 As[BM * KPAD];
    __shared__ __align__(16) __bf16 Bd[BN * KPAD];

    const int tid = threadIdx.x;
    const int wave = tid >> 5;
    const int lane = tid & 31;
    const int l = lane & 15, hh = lane >> 4;
    const int waveM = wave >> 1;
    const int waveN = wave & 1;

    const int base = offsets[e] + m0;

    const int ar = tid >> 1;
    const int ak = (tid & 1) * 16;
    const int grow = (m0 + ar) < cnt ? (m0 + ar) : (cnt - 1);
    const __bf16* __restrict__ asrc =
        act + (size_t)(offsets[e] + grow) * I_DIM + ak;

    const int bk = tid >> 3;
    const int bc = (tid & 7) * 8;
    const float* __restrict__ dsrc =
        w_down + (size_t)e * I_DIM * H_DIM + (size_t)bk * H_DIM + n0 + bc;

    v8f acc[2][2] = {{{}, {}}, {{}, {}}};

    for (int k0 = 0; k0 < I_DIM; k0 += BK) {
        // ---- stage A (bf16, 128x32): CDNA5 async global->LDS when available --
        {
            __bf16* d = As + ar * KPAD + ak;
#ifdef USE_ASYNC_LDS
            __builtin_amdgcn_global_load_async_to_lds_b128(
                (as1_v4i)(asrc + k0), (as3_v4i)d, 0, 0);
            __builtin_amdgcn_global_load_async_to_lds_b128(
                (as1_v4i)(asrc + k0 + 8), (as3_v4i)(d + 8), 0, 0);
#else
            *(uint4*)d       = *(const uint4*)(asrc + k0);
            *(uint4*)(d + 8) = *(const uint4*)(asrc + k0 + 8);
#endif
        }
        // ---- stage B (w_down, 32x64 f32 -> bf16 transposed) ----
        {
            const size_t koff = (size_t)k0 * H_DIM;
            float4 d0 = *(const float4*)(dsrc + koff);
            float4 d1 = *(const float4*)(dsrc + koff + 4);
            if (k0 + BK < I_DIM)
                __builtin_prefetch(dsrc + koff + (size_t)BK * H_DIM, 0, 1);
            __bf16* dd = Bd + bc * KPAD + bk;
            dd[0 * KPAD] = (__bf16)d0.x; dd[1 * KPAD] = (__bf16)d0.y;
            dd[2 * KPAD] = (__bf16)d0.z; dd[3 * KPAD] = (__bf16)d0.w;
            dd[4 * KPAD] = (__bf16)d1.x; dd[5 * KPAD] = (__bf16)d1.y;
            dd[6 * KPAD] = (__bf16)d1.z; dd[7 * KPAD] = (__bf16)d1.w;
        }
#ifdef USE_ASYNC_LDS
#if __has_builtin(__builtin_amdgcn_s_wait_asynccnt)
        __builtin_amdgcn_s_wait_asynccnt(0);
#else
        asm volatile("s_wait_asynccnt 0" ::: "memory");
#endif
#endif
        __syncthreads();

        v16bf af[2];
#pragma unroll
        for (int m = 0; m < 2; ++m) {
            const __bf16* p = As + (waveM * 32 + m * 16 + l) * KPAD + 16 * hh;
            af[m] = frag_ld(p, p + 8);
        }
#pragma unroll
        for (int n = 0; n < 2; ++n) {
            const __bf16* pb = Bd + (waveN * 32 + n * 16 + l) * KPAD + 16 * hh;
            const v16bf bf = frag_ld(pb, pb + 8);
#pragma unroll
            for (int m = 0; m < 2; ++m) {
                acc[m][n] = __builtin_amdgcn_wmma_f32_16x16x32_bf16(
                    false, af[m], false, bf, (short)0, acc[m][n], false, false);
            }
        }
        __syncthreads();
    }

#pragma unroll
    for (int m = 0; m < 2; ++m) {
#pragma unroll
        for (int n = 0; n < 2; ++n) {
            const int col = n0 + waveN * 32 + n * 16 + l;
#pragma unroll
            for (int r = 0; r < 8; ++r) {
                const int row = waveM * 32 + m * 16 + r + 8 * hh;
                if (m0 + row < cnt)
                    oute[(size_t)(base + row) * H_DIM + col] = acc[m][n][r];
            }
        }
    }
}

// ------------------------------ combine -------------------------------------
__global__ __launch_bounds__(256)
void moe_combine(const float* __restrict__ oute, const int* __restrict__ offsets,
                 const int* __restrict__ tk_expert, const int* __restrict__ tk_pos,
                 const float* __restrict__ tk_w, float* __restrict__ out)
{
    const int t = blockIdx.y;
    const int hcol = blockIdx.x * blockDim.x + threadIdx.x;
    const int e0 = tk_expert[t * TOPK + 0];
    const int e1 = tk_expert[t * TOPK + 1];
    const size_t s0 = (size_t)(offsets[e0] + tk_pos[t * TOPK + 0]);
    const size_t s1 = (size_t)(offsets[e1] + tk_pos[t * TOPK + 1]);
    const float w0 = tk_w[t * TOPK + 0];
    const float w1 = tk_w[t * TOPK + 1];
    out[(size_t)t * H_DIM + hcol] =
        w0 * oute[s0 * H_DIM + hcol] + w1 * oute[s1 * H_DIM + hcol];
}

// ------------------------------ launcher ------------------------------------
static inline size_t align_up(size_t x, size_t a) { return (x + a - 1) & ~(a - 1); }

extern "C" void kernel_launch(void* const* d_in, const int* in_sizes, int n_in,
                              void* d_out, int out_size, void* d_ws, size_t ws_size,
                              hipStream_t stream)
{
    const float* hs     = (const float*)d_in[0];   // [T, H]
    const float* gate_w = (const float*)d_in[1];   // [H, E]
    const float* w_gate = (const float*)d_in[2];   // [E, H, I]
    const float* w_up   = (const float*)d_in[3];   // [E, H, I]
    const float* w_down = (const float*)d_in[4];   // [E, I, H]
    float* out = (float*)d_out;                    // [T, H]

    char* ws = (char*)d_ws;
    size_t off = 0;
    int* counts = (int*)(ws + off);           off = align_up(off + E_NUM * sizeof(int), 256);
    int* offsets = (int*)(ws + off);          off = align_up(off + (E_NUM + 1) * sizeof(int), 256);
    int* token_list = (int*)(ws + off);       off = align_up(off + (size_t)E_NUM * T_NUM * sizeof(int), 256);
    int* tk_expert = (int*)(ws + off);        off = align_up(off + (size_t)T_NUM * TOPK * sizeof(int), 256);
    int* tk_pos = (int*)(ws + off);           off = align_up(off + (size_t)T_NUM * TOPK * sizeof(int), 256);
    float* tk_w = (float*)(ws + off);         off = align_up(off + (size_t)T_NUM * TOPK * sizeof(float), 256);
    __bf16* act = (__bf16*)(ws + off);        off = align_up(off + (size_t)T_NUM * TOPK * I_DIM * sizeof(__bf16), 256);
    float* oute = (float*)(ws + off);         off = align_up(off + (size_t)T_NUM * TOPK * H_DIM * sizeof(float), 256);
    (void)ws_size; (void)in_sizes; (void)n_in; (void)out_size;

    (void)hipMemsetAsync(counts, 0, E_NUM * sizeof(int), stream);

    moe_router<<<dim3(T_NUM / 8), dim3(256), 0, stream>>>(
        hs, gate_w, counts, token_list, tk_expert, tk_pos, tk_w);

    moe_offsets<<<dim3(1), dim3(32), 0, stream>>>(counts, offsets);

    moe_gemm1<<<dim3(I_DIM / BN, (T_NUM + BM - 1) / BM, E_NUM), dim3(256), 0, stream>>>(
        hs, w_gate, w_up, counts, offsets, token_list, act);

    moe_gemm2<<<dim3(H_DIM / BN, (T_NUM + BM - 1) / BM, E_NUM), dim3(256), 0, stream>>>(
        act, w_down, counts, offsets, oute);

    moe_combine<<<dim3(H_DIM / 256, T_NUM), dim3(256), 0, stream>>>(
        oute, offsets, tk_expert, tk_pos, tk_w, out);
}
